// Set2SetPooling_57655640982226
// MI455X (gfx1250) — matrix operand order repaired and checked by reading, compiled, not run on hardware
//
#include <hip/hip_runtime.h>
#include <hip/hip_bf16.h>

// ---------------------------------------------------------------------------
// Set2Set pooling for MI455X (gfx1250, wave32, WMMA).
//   step x3: gates = Xb @ [W_ih|W_hh]^T + (b_ih+b_hh)    (bf16 WMMA, K=1536)
//            LSTM pointwise -> h (f32) and Xb (bf16)
//            segment softmax attention over x            -> r into Xb
//   out = Xb[:, :1024] @ W_proj^T + b_proj               (bf16 WMMA, K=1024)
// Xb[256,1536] = [ h(512) | r(512) | h(512) ]  -- bf16, single activation buf
// ---------------------------------------------------------------------------

#define NNODES   100000
#define DDIM     512
#define NGRAPHS  256
#define GATE4    2048      // 4*d
#define QDIM     1024      // 2*d
#define KGATES   1536      // 2*d + d
#define DOUT     256

typedef __attribute__((ext_vector_type(16))) __bf16 v16bf;
typedef __attribute__((ext_vector_type(8)))  __bf16 v8bf;
typedef __attribute__((ext_vector_type(8)))  float  v8f;

// ---------------------------------------------------------------------------
// Prep: bf16 weight conversion (combined [W_ih|W_hh] row-major), bias fuse,
// zero-init of h/c/Xb. One element per thread.
// ---------------------------------------------------------------------------
__global__ __launch_bounds__(256)
void prep_kernel(const float* __restrict__ W_ih, const float* __restrict__ W_hh,
                 const float* __restrict__ b_ih, const float* __restrict__ b_hh,
                 const float* __restrict__ W_proj,
                 __bf16* __restrict__ Wc, __bf16* __restrict__ Wp,
                 float* __restrict__ bias_g,
                 float* __restrict__ h, float* __restrict__ c,
                 __bf16* __restrict__ Xb) {
    long long idx = (long long)blockIdx.x * 256 + threadIdx.x;
    const long long T_WC = (long long)GATE4 * KGATES;    // 3,145,728
    const long long T_WP = (long long)DOUT * QDIM;       //   262,144
    const long long T_B  = GATE4;                        //     2,048
    const long long T_H  = (long long)NGRAPHS * DDIM;    //   131,072 (h and c)
    const long long T_X  = (long long)NGRAPHS * KGATES;  //   393,216

    if (idx < T_WC) {
        int n = (int)(idx / KGATES), k = (int)(idx % KGATES);
        float v = (k < QDIM) ? W_ih[(size_t)n * QDIM + k]
                             : W_hh[(size_t)n * DDIM + (k - QDIM)];
        Wc[idx] = (__bf16)v;
        return;
    }
    idx -= T_WC;
    if (idx < T_WP) { Wp[idx] = (__bf16)W_proj[idx]; return; }
    idx -= T_WP;
    if (idx < T_B)  { bias_g[idx] = b_ih[idx] + b_hh[idx]; return; }
    idx -= T_B;
    if (idx < T_H)  { h[idx] = 0.f; c[idx] = 0.f; return; }
    idx -= T_H;
    if (idx < T_X)  { Xb[idx] = (__bf16)0.f; return; }
}

// ---------------------------------------------------------------------------
// bf16 WMMA GEMM: C[M,N] = A[M,Ktot](lda) @ W[N,Ktot]^T + bias[N]
// A and W both pre-converted bf16 row-major. One wave = one 16x16 C tile,
// 8 waves/block. Inner loop: 2x b128 (A runs) + 2x b128 (B runs) + 1 WMMA.
// ---------------------------------------------------------------------------
__global__ __launch_bounds__(256)
void gemm_wmma_bf16(const __bf16* __restrict__ A, int lda,
                    const __bf16* __restrict__ W, const float* __restrict__ bias,
                    float* __restrict__ C, int N, int Ktot) {
    const int lane   = threadIdx.x & 31;
    const int wave   = threadIdx.x >> 5;
    const int tilesN = N >> 4;
    const int tile   = blockIdx.x * 8 + wave;
    const int tm     = (tile / tilesN) << 4;
    const int tn     = (tile % tilesN) << 4;
    const int lo     = lane & 15;
    const int hi     = lane >> 4;

    const __bf16* arow = A + (size_t)(tm + lo) * lda;    // A row this lane feeds
    const __bf16* wrow = W + (size_t)(tn + lo) * Ktot;   // B column this lane owns

    v8f acc = {0.f, 0.f, 0.f, 0.f, 0.f, 0.f, 0.f, 0.f};

    for (int k0 = 0; k0 < Ktot; k0 += 32) {
        // 16-bit A layout: lane holds two contiguous runs of 8 K-values:
        //   K = k0+hi*8 + (0..7)  and  K = k0+16+hi*8 + (0..7)
        v8bf alo = *(const v8bf*)(arow + k0 + hi * 8);
        v8bf ahi = *(const v8bf*)(arow + k0 + 16 + hi * 8);
        v16bf a  = __builtin_shufflevector(alo, ahi,
                       0, 1, 2, 3, 4, 5, 6, 7, 8, 9, 10, 11, 12, 13, 14, 15);
        // B layout: lane = column, K = k0+hi*8 + (0..7) and k0+16+hi*8 + (0..7)
        v8bf blo = *(const v8bf*)(wrow + k0 + hi * 8);
        v8bf bhi = *(const v8bf*)(wrow + k0 + 16 + hi * 8);
        v16bf b  = __builtin_shufflevector(blo, bhi,
                       0, 1, 2, 3, 4, 5, 6, 7, 8, 9, 10, 11, 12, 13, 14, 15);

        acc = __builtin_amdgcn_wmma_f32_16x16x32_bf16(
                  false, a, false, b, (short)0, acc, false, false);
    }

    const float bn   = bias ? bias[tn + lo] : 0.f;
    const int mBase  = tm + hi * 8;   // C layout: lanes 0-15 -> M rows 0..7, 16-31 -> +8
#pragma unroll
    for (int v = 0; v < 8; ++v)
        C[(size_t)(mBase + v) * N + (tn + lo)] = acc[v] + bn;
}

// ---------------------------------------------------------------------------
// LSTM pointwise (PyTorch gate order i,f,g,o). Writes f32 h (for attention)
// and bf16 h into Xb columns [0,512) and [1024,1536) (GEMM operand).
// ---------------------------------------------------------------------------
__global__ __launch_bounds__(256)
void lstm_cell(const float* __restrict__ gates, float* __restrict__ h,
               float* __restrict__ c, __bf16* __restrict__ Xb) {
    int idx = blockIdx.x * 256 + threadIdx.x;
    if (idx >= NGRAPHS * DDIM) return;
    int b = idx >> 9;          // / 512
    int j = idx & (DDIM - 1);
    const float* g = gates + (size_t)b * GATE4;
    float si = 1.f / (1.f + __expf(-g[j]));
    float sf = 1.f / (1.f + __expf(-g[j + DDIM]));
    float tg = tanhf(g[j + 2 * DDIM]);
    float so = 1.f / (1.f + __expf(-g[j + 3 * DDIM]));
    float cn = sf * c[idx] + si * tg;
    float hn = so * tanhf(cn);
    c[idx] = cn;
    h[idx] = hn;
    __bf16 hb = (__bf16)hn;
    __bf16* xrow = Xb + (size_t)b * KGATES;
    xrow[j]        = hb;       // q_star[:, :d]
    xrow[QDIM + j] = hb;       // h operand of W_hh block
}

// ---------------------------------------------------------------------------
// Segment softmax attention. One block (8 waves) per graph; batch is sorted,
// segment bounds via binary search. Wave-per-node dot products (lane owns 16
// dims, float4 loads, shfl_xor reduce), e cached in global scratch,
// cross-wave reductions in LDS. The r-sweep re-reads this graph's ~800KB
// x slice from L2. r written as bf16 into Xb columns [512,1024).
// ---------------------------------------------------------------------------
__global__ __launch_bounds__(256)
void set2set_attention(const float* __restrict__ x, const int* __restrict__ batch,
                       const float* __restrict__ h, float* __restrict__ e_buf,
                       __bf16* __restrict__ Xb) {
    const int b    = blockIdx.x;
    const int lane = threadIdx.x & 31;
    const int wave = threadIdx.x >> 5;

    // lower_bound(batch, b) and lower_bound(batch, b+1)
    int lo_ = 0, hi_ = NNODES;
    while (lo_ < hi_) { int mid = (lo_ + hi_) >> 1; if (batch[mid] < b) lo_ = mid + 1; else hi_ = mid; }
    const int start = lo_;
    hi_ = NNODES;
    while (lo_ < hi_) { int mid = (lo_ + hi_) >> 1; if (batch[mid] < b + 1) lo_ = mid + 1; else hi_ = mid; }
    const int end = lo_;

    // query h[b]; lane owns dims [lane*16, lane*16+16)
    const float* hb = h + (size_t)b * DDIM + lane * 16;
    float4 hq0 = ((const float4*)hb)[0];
    float4 hq1 = ((const float4*)hb)[1];
    float4 hq2 = ((const float4*)hb)[2];
    float4 hq3 = ((const float4*)hb)[3];

    __shared__ float red[8];
    __shared__ float rbuf[8][DDIM];

    // ---- pass 1: e[n] = <x[n], h[b]>, segment max ----
    float mloc = -__builtin_inff();
    for (int nn = start + wave; nn < end; nn += 8) {
        const float* xr = x + (size_t)nn * DDIM + lane * 16;
        __builtin_prefetch(xr + 8 * DDIM, 0, 1);   // -> global_prefetch_b8
        float4 v0 = ((const float4*)xr)[0];
        float4 v1 = ((const float4*)xr)[1];
        float4 v2 = ((const float4*)xr)[2];
        float4 v3 = ((const float4*)xr)[3];
        float s = v0.x*hq0.x + v0.y*hq0.y + v0.z*hq0.z + v0.w*hq0.w
                + v1.x*hq1.x + v1.y*hq1.y + v1.z*hq1.z + v1.w*hq1.w
                + v2.x*hq2.x + v2.y*hq2.y + v2.z*hq2.z + v2.w*hq2.w
                + v3.x*hq3.x + v3.y*hq3.y + v3.z*hq3.z + v3.w*hq3.w;
#pragma unroll
        for (int off = 16; off > 0; off >>= 1) s += __shfl_xor(s, off, 32);
        if (lane == 0) e_buf[nn] = s;
        mloc = fmaxf(mloc, s);
    }
    if (lane == 0) red[wave] = mloc;
    __syncthreads();
    float mseg = -__builtin_inff();
#pragma unroll
    for (int w = 0; w < 8; ++w) mseg = fmaxf(mseg, red[w]);
    if (!(mseg > -1e30f)) mseg = 0.f;     // guard empty segments
    __syncthreads();

    // ---- pass 2: denom = sum exp(e - m) ----
    float dloc = 0.f;
    for (int nn = start + (int)threadIdx.x; nn < end; nn += 256)
        dloc += __expf(e_buf[nn] - mseg);
#pragma unroll
    for (int off = 16; off > 0; off >>= 1) dloc += __shfl_xor(dloc, off, 32);
    if (lane == 0) red[wave] = dloc;
    __syncthreads();
    float denom = 1e-16f;
#pragma unroll
    for (int w = 0; w < 8; ++w) denom += red[w];
    const float inv = 1.f / denom;

    // ---- pass 3: r = sum a[n] * x[n]  (x slice re-read from L2) ----
    float racc[16];
#pragma unroll
    for (int t = 0; t < 16; ++t) racc[t] = 0.f;
    for (int nn = start + wave; nn < end; nn += 8) {
        float aw = __expf(e_buf[nn] - mseg) * inv;
        const float* xr = x + (size_t)nn * DDIM + lane * 16;
        float4 v0 = ((const float4*)xr)[0];
        float4 v1 = ((const float4*)xr)[1];
        float4 v2 = ((const float4*)xr)[2];
        float4 v3 = ((const float4*)xr)[3];
        racc[0]  += aw * v0.x; racc[1]  += aw * v0.y; racc[2]  += aw * v0.z; racc[3]  += aw * v0.w;
        racc[4]  += aw * v1.x; racc[5]  += aw * v1.y; racc[6]  += aw * v1.z; racc[7]  += aw * v1.w;
        racc[8]  += aw * v2.x; racc[9]  += aw * v2.y; racc[10] += aw * v2.z; racc[11] += aw * v2.w;
        racc[12] += aw * v3.x; racc[13] += aw * v3.y; racc[14] += aw * v3.z; racc[15] += aw * v3.w;
    }
#pragma unroll
    for (int t = 0; t < 16; ++t) rbuf[wave][lane * 16 + t] = racc[t];
    __syncthreads();
    for (int jd = threadIdx.x; jd < DDIM; jd += 256) {
        float s = 0.f;
#pragma unroll
        for (int w = 0; w < 8; ++w) s += rbuf[w][jd];
        Xb[(size_t)b * KGATES + DDIM + jd] = (__bf16)s;   // q_star[:, d:2d]
    }
}

// ---------------------------------------------------------------------------
extern "C" void kernel_launch(void* const* d_in, const int* in_sizes, int n_in,
                              void* d_out, int out_size, void* d_ws, size_t ws_size,
                              hipStream_t stream) {
    const float* x      = (const float*)d_in[0];
    const int*   batch  = (const int*)  d_in[1];
    const float* W_ih   = (const float*)d_in[2];
    const float* W_hh   = (const float*)d_in[3];
    const float* b_ih   = (const float*)d_in[4];
    const float* b_hh   = (const float*)d_in[5];
    const float* W_proj = (const float*)d_in[6];
    const float* b_proj = (const float*)d_in[7];
    float*       out    = (float*)d_out;

    char* ws = (char*)d_ws;
    size_t off = 0;
    auto take = [&](size_t bytes) -> char* {
        char* p = ws + off;
        off += (bytes + 255) & ~(size_t)255;
        return p;
    };
    __bf16* Wc     = (__bf16*)take((size_t)GATE4 * KGATES * 2);    // 6.29 MB
    __bf16* Wp     = (__bf16*)take((size_t)DOUT * QDIM * 2);       // 0.52 MB
    __bf16* Xb     = (__bf16*)take((size_t)NGRAPHS * KGATES * 2);  // 0.79 MB
    float*  bias_g = (float*) take((size_t)GATE4 * 4);
    float*  h      = (float*) take((size_t)NGRAPHS * DDIM * 4);
    float*  c      = (float*) take((size_t)NGRAPHS * DDIM * 4);
    float*  gates  = (float*) take((size_t)NGRAPHS * GATE4 * 4);
    float*  e_buf  = (float*) take((size_t)NNODES * 4);

    // prep: weights->bf16, bias fuse, zero state
    long long prep_total = (long long)GATE4 * KGATES + (long long)DOUT * QDIM
                         + GATE4 + (long long)NGRAPHS * DDIM
                         + (long long)NGRAPHS * KGATES;
    int prep_blocks = (int)((prep_total + 255) / 256);
    prep_kernel<<<prep_blocks, 256, 0, stream>>>(W_ih, W_hh, b_ih, b_hh, W_proj,
                                                 Wc, Wp, bias_g, h, c, Xb);

    for (int step = 0; step < 3; ++step) {
        // gates[256,2048] = Xb @ Wc^T + bias : (16x128) tiles / 8 waves
        gemm_wmma_bf16<<<256, 256, 0, stream>>>(Xb, KGATES, Wc, bias_g,
                                                gates, GATE4, KGATES);
        lstm_cell<<<(NGRAPHS * DDIM) / 256, 256, 0, stream>>>(gates, h, c, Xb);
        set2set_attention<<<NGRAPHS, 256, 0, stream>>>(x, batch, h, e_buf, Xb);
    }

    // out[256,256] = Xb[:, :1024] @ W_proj^T + b_proj : (16x16) tiles / 8 waves
    gemm_wmma_bf16<<<32, 256, 0, stream>>>(Xb, KGATES, Wp, b_proj,
                                           out, DOUT, QDIM);
}
